// DGCNNSeg_31387620999470
// MI455X (gfx1250) — matrix operand order repaired and checked by reading, not compiled, not run on hardware
//
#include <hip/hip_runtime.h>
#include <hip/hip_bf16.h>

// ---------------------------------------------------------------------------
// DGCNN segmentation forward for MI455X (gfx1250).
// Matmuls: v_wmma_f32_16x16x32_f16, one wave per 16x64 output strip
// (A-fragment reused across 4 N-tiles). K-loop is manually unrolled 2x with
// two distinct load-buffer sets so the double buffer swaps roles by
// unrolling (no v_mov rotation, no WMMA->VALU WAR hazard nops).
// Streaming pack/gather kernels are element-per-thread for coalescing
// (workload is HBM-bound: ~0.5-1 GB intermediates vs ~50 GFLOP).
// ---------------------------------------------------------------------------

#define EPSV   1e-5f
#define SLOPEV 0.2f

constexpr int B_  = 4;
constexpr int N_  = 4096;
constexpr int KNN = 20;
constexpr int EM  = B_ * N_ * KNN;   // 327680 edge rows
constexpr int PM  = B_ * N_;         // 16384 point rows

typedef __attribute__((ext_vector_type(16))) _Float16 v16h;
typedef __attribute__((ext_vector_type(8)))  _Float16 v8h;
typedef __attribute__((ext_vector_type(8)))  float    v8f;

// ---------------------------------------------------------------------------
// WMMA GEMM:  D[M, Nc] = A[M, Kp] x W[Nc, Kp]^T   (A, W f16, D f32)
// M mult of 16, Nc mult of 16*NT, Kp mult of 32.
// ---------------------------------------------------------------------------
template <int NT>
__global__ __launch_bounds__(128) void wmma_gemm_nt(
    const _Float16* __restrict__ A, const _Float16* __restrict__ W,
    float* __restrict__ D, int M, int Nc, int Kp) {
  const int wave = threadIdx.x >> 5;
  const int lane = threadIdx.x & 31;
  const int nstrips = Nc / (16 * NT);
  const int strips  = (M >> 4) * nstrips;
  int s = blockIdx.x * 4 + wave;
  if (s >= strips) return;              // wave-uniform: EXEC stays full
  const int tm = s / nstrips;
  const int tnBase = (s % nstrips) * NT;
  const int half = lane >> 4;           // 0 or 1
  const int l16  = lane & 15;

  // A fragment layout (matches B): lane half 0 holds K[k..k+15];
  // half 1 holds K[k+16..k+31].
  const _Float16* __restrict__ Arow = A + (size_t)(tm * 16 + l16) * Kp + half * 16;
  // B fragment: lane holds weight row (output channel) tnBase*16+t*16+l16,
  // 16 contiguous K starting at k + half*16.
  const _Float16* __restrict__ Wb   = W + (size_t)(tnBase * 16 + l16) * Kp + half * 16;

  const v8f vzero = {};
  v8f acc[NT];
  #pragma unroll
  for (int t = 0; t < NT; ++t) acc[t] = vzero;

  // set-0 buffers (loop-carried, single def per path -> no phi copies)
  v8h a0 = *(const v8h*)(Arow);
  v8h a1 = *(const v8h*)(Arow + 8);
  v16h b0[NT];
  #pragma unroll
  for (int t = 0; t < NT; ++t) b0[t] = *(const v16h*)(Wb + (size_t)t * 16 * Kp);

  int k = 0;
  while (k + 32 < Kp) {                 // >= 2 chunks remain
    // load chunk k+32 into set-1 (loop-local SSA)
    v8h c0 = *(const v8h*)(Arow + k + 32);
    v8h c1 = *(const v8h*)(Arow + k + 40);
    v16h b1[NT];
    #pragma unroll
    for (int t = 0; t < NT; ++t)
      b1[t] = *(const v16h*)(Wb + (size_t)t * 16 * Kp + k + 32);
    __builtin_prefetch(Arow + k + 96, 0, 3);     // global_prefetch_b8

    // compute chunk k on set-0
    {
      v16h a = __builtin_shufflevector(a0, a1,
          0,1,2,3,4,5,6,7,8,9,10,11,12,13,14,15);
      #pragma unroll
      for (int t = 0; t < NT; ++t)
        acc[t] = __builtin_amdgcn_wmma_f32_16x16x32_f16(
            false, a, false, b0[t], (short)0, acc[t], false, false);
    }
    k += 32;

    if (k + 32 < Kp) {                  // refill set-0 with chunk k+32
      a0 = *(const v8h*)(Arow + k + 32);
      a1 = *(const v8h*)(Arow + k + 40);
      #pragma unroll
      for (int t = 0; t < NT; ++t)
        b0[t] = *(const v16h*)(Wb + (size_t)t * 16 * Kp + k + 32);
    }

    // compute chunk k on set-1
    {
      v16h a = __builtin_shufflevector(c0, c1,
          0,1,2,3,4,5,6,7,8,9,10,11,12,13,14,15);
      #pragma unroll
      for (int t = 0; t < NT; ++t)
        acc[t] = __builtin_amdgcn_wmma_f32_16x16x32_f16(
            false, a, false, b1[t], (short)0, acc[t], false, false);
    }
    k += 32;
  }
  if (k < Kp) {                         // odd chunk count tail (set-0 holds it)
    v16h a = __builtin_shufflevector(a0, a1,
        0,1,2,3,4,5,6,7,8,9,10,11,12,13,14,15);
    #pragma unroll
    for (int t = 0; t < NT; ++t)
      acc[t] = __builtin_amdgcn_wmma_f32_16x16x32_f16(
          false, a, false, b0[t], (short)0, acc[t], false, false);
  }

  // C/D layout: acc[t][r] = D[tm*16 + half*8 + r][tnBase*16 + t*16 + l16]
  float* __restrict__ Drow = D + (size_t)(tm * 16 + half * 8) * Nc + tnBase * 16 + l16;
  #pragma unroll
  for (int t = 0; t < NT; ++t) {
    #pragma unroll
    for (int r = 0; r < 8; ++r) Drow[(size_t)r * Nc + t * 16] = acc[t][r];
  }
}

// ---------------------------------------------------------------------------
// Weight convert + pad: W f32 [co, ci] -> f16 [coP, ciP] (zero padded)
// ---------------------------------------------------------------------------
__global__ void pack_w(const float* __restrict__ W, _Float16* __restrict__ Wh,
                       int co, int ci, int coP, int ciP) {
  int t = blockIdx.x * blockDim.x + threadIdx.x;
  if (t >= coP * ciP) return;
  int o = t / ciP, c = t % ciP;
  Wh[t] = (o < co && c < ci) ? (_Float16)W[(size_t)o * ci + c] : (_Float16)0.f;
}

// f32 activations [rows, C] -> f16 [rows, Kp] zero padded (element/thread)
__global__ void pack_rows(const float* __restrict__ A, _Float16* __restrict__ E,
                          int rows, int C, int Kp) {
  size_t id = (size_t)blockIdx.x * blockDim.x + threadIdx.x;
  if (id >= (size_t)rows * Kp) return;
  int c = (int)(id % Kp);
  int r = (int)(id / Kp);
  E[id] = (_Float16)((c < C) ? A[(size_t)r * C + c] : 0.f);
}

// ---------------------------------------------------------------------------
// kNN: one thread per point; insertion-sorted top-(K+1) smallest distances,
// drop nearest (self). x is [B, N, C] row-major.
// ---------------------------------------------------------------------------
template <int C>
__global__ void knn_kernel(const float* __restrict__ x, int* __restrict__ idx) {
  int gid = blockIdx.x * blockDim.x + threadIdx.x;
  if (gid >= B_ * N_) return;
  int b = gid / N_, i = gid % N_;
  const float* xb = x + (size_t)b * N_ * C;
  float xi[C];
  #pragma unroll
  for (int c = 0; c < C; ++c) xi[c] = xb[(size_t)i * C + c];
  const int KK = KNN + 1;
  float bd[KK]; int bi[KK];
  for (int t = 0; t < KK; ++t) { bd[t] = 3.4e38f; bi[t] = 0; }
  for (int j = 0; j < N_; ++j) {
    const float* xj = xb + (size_t)j * C;
    float d = 0.f;
    #pragma unroll
    for (int c = 0; c < C; ++c) { float t = xi[c] - xj[c]; d += t * t; }
    if (d < bd[KK - 1]) {
      int p = KK - 1;
      while (p > 0 && bd[p - 1] > d) { bd[p] = bd[p - 1]; bi[p] = bi[p - 1]; --p; }
      bd[p] = d; bi[p] = j;
    }
  }
  int* out = idx + (size_t)gid * KNN;
  for (int t = 0; t < KNN; ++t) out[t] = bi[t + 1];   // skip self
}

// Edge features: E[row, :] = [x[nb]-x[i] | x[i]] padded to Kp, f16 (elem/thread)
__global__ void gather_edge(const float* __restrict__ x, const int* __restrict__ idx,
                            _Float16* __restrict__ E, int C, int Kp) {
  size_t id = (size_t)blockIdx.x * blockDim.x + threadIdx.x;
  if (id >= (size_t)EM * Kp) return;
  int c = (int)(id % Kp);
  int r = (int)(id / Kp);
  float val = 0.f;
  if (c < 2 * C) {
    int j = r % KNN, pn = r / KNN, b = pn / N_;
    const float* xc = x + (size_t)pn * C;
    if (c < C) {
      int nb = idx[(size_t)pn * KNN + j];
      const float* xn = x + ((size_t)b * N_ + nb) * C;
      val = xn[c] - xc[c];
    } else {
      val = xc[c - C];
    }
  }
  E[id] = (_Float16)val;
}

// ---------------------------------------------------------------------------
// BatchNorm: pass 1 per-column mean/var via LDS tree reduce (block per column)
// ---------------------------------------------------------------------------
__global__ void bn_stats(const float* __restrict__ Y, float* __restrict__ stats,
                         int M, int Nc) {
  __shared__ float ss[256], sq[256];
  int col = blockIdx.x;
  float a = 0.f, b = 0.f;
  for (int r = threadIdx.x; r < M; r += blockDim.x) {
    float v = Y[(size_t)r * Nc + col];
    a += v; b += v * v;
  }
  ss[threadIdx.x] = a; sq[threadIdx.x] = b;
  __syncthreads();
  for (int s = blockDim.x >> 1; s > 0; s >>= 1) {
    if ((int)threadIdx.x < s) {
      ss[threadIdx.x] += ss[threadIdx.x + s];
      sq[threadIdx.x] += sq[threadIdx.x + s];
    }
    __syncthreads();
  }
  if (threadIdx.x == 0) {
    float m = ss[0] / (float)M;
    stats[col] = m;
    stats[Nc + col] = fmaxf(sq[0] / (float)M - m * m, 0.f);
  }
}

// pass 2: in-place normalize + affine + LeakyReLU
__global__ void bn_apply_leaky(float* __restrict__ Y, const float* __restrict__ stats,
                               const float* __restrict__ g, const float* __restrict__ bb,
                               int M, int Nc, int NcReal) {
  size_t t = (size_t)blockIdx.x * blockDim.x + threadIdx.x;
  if (t >= (size_t)M * Nc) return;
  int col = (int)(t % Nc);
  if (col >= NcReal) return;
  float m = stats[col], v = stats[Nc + col];
  float y = Y[t];
  y = (y - m) * (g[col] * rsqrtf(v + EPSV)) + bb[col];
  Y[t] = (y >= 0.f) ? y : SLOPEV * y;
}

// max over neighbor axis: [P, KNN, C] -> [P, C]
__global__ void max_over_k(const float* __restrict__ Y, float* __restrict__ O, int C) {
  int t = blockIdx.x * blockDim.x + threadIdx.x;
  if (t >= PM * C) return;
  int p = t / C, c = t % C;
  const float* y = Y + ((size_t)p * KNN) * C + c;
  float m = y[0];
  for (int j = 1; j < KNN; ++j) m = fmaxf(m, y[(size_t)j * C]);
  O[(size_t)p * C + c] = m;
}

// max over points: [B, N, C] -> [B, C]
__global__ void max_over_n(const float* __restrict__ Y, float* __restrict__ O, int C) {
  int t = blockIdx.x * blockDim.x + threadIdx.x;
  if (t >= B_ * C) return;
  int b = t / C, c = t % C;
  const float* y = Y + (size_t)b * N_ * C + c;
  float m = y[0];
  for (int n = 1; n < N_; ++n) m = fmaxf(m, y[(size_t)n * C]);
  O[t] = m;
}

// tiny dense layer + BN over batch + leaky (B=4 rows; one thread per channel)
__global__ void lin_bn_leaky(const float* __restrict__ X, const float* __restrict__ W,
                             const float* __restrict__ bias, const float* __restrict__ g,
                             const float* __restrict__ bb, float* __restrict__ O,
                             int cin, int cout) {
  int o = blockIdx.x * blockDim.x + threadIdx.x;
  if (o >= cout) return;
  float y[B_]; float m = 0.f;
  for (int b = 0; b < B_; ++b) {
    const float* xr = X + (size_t)b * cin;
    const float* wr = W + (size_t)o * cin;
    float s = bias[o];
    for (int c = 0; c < cin; ++c) s += xr[c] * wr[c];
    y[b] = s; m += s;
  }
  m *= (1.f / B_);
  float v = 0.f;
  for (int b = 0; b < B_; ++b) { float d = y[b] - m; v += d * d; }
  v *= (1.f / B_);
  float sc = g[o] * rsqrtf(v + EPSV);
  for (int b = 0; b < B_; ++b) {
    float z = (y[b] - m) * sc + bb[o];
    O[(size_t)b * cout + o] = (z >= 0.f) ? z : SLOPEV * z;
  }
}

// transform matrix: mat[b, 9] = t[b,256] . W[9,256]^T + bias[9]
__global__ void transform_mat(const float* __restrict__ t, const float* __restrict__ W,
                              const float* __restrict__ bias, float* __restrict__ mat) {
  int id = blockIdx.x * blockDim.x + threadIdx.x;
  if (id >= B_ * 9) return;
  int b = id / 9, o = id % 9;
  float s = bias[o];
  for (int c = 0; c < 256; ++c) s += t[(size_t)b * 256 + c] * W[(size_t)o * 256 + c];
  mat[id] = s;
}

// xS[b,n,d] = sum_c xT[b,n,c] * mat[b][c*3+d]
__global__ void apply_transform(const float* __restrict__ xT, const float* __restrict__ mat,
                                float* __restrict__ xS) {
  int id = blockIdx.x * blockDim.x + threadIdx.x;
  if (id >= B_ * N_) return;
  int b = id / N_;
  const float* xi = xT + (size_t)id * 3;
  const float* m = mat + (size_t)b * 9;
  float* o = xS + (size_t)id * 3;
  #pragma unroll
  for (int d = 0; d < 3; ++d)
    o[d] = xi[0] * m[d] + xi[1] * m[3 + d] + xi[2] * m[6 + d];
}

// input transpose (B,3,N) -> (B,N,3)
__global__ void transpose_in(const float* __restrict__ x, float* __restrict__ xT) {
  int id = blockIdx.x * blockDim.x + threadIdx.x;
  if (id >= B_ * N_ * 3) return;
  int c = id % 3, p = id / 3;
  int b = p / N_, n = p % N_;
  xT[id] = x[((size_t)b * 3 + c) * N_ + n];
}

// concat x1|x2|x3 (each [P,64]) -> f16 [P,192] (element/thread)
__global__ void pack_concat3(const float* __restrict__ x1, const float* __restrict__ x2,
                             const float* __restrict__ x3, _Float16* __restrict__ E) {
  int id = blockIdx.x * blockDim.x + threadIdx.x;
  if (id >= PM * 192) return;
  int c = id % 192, p = id / 192;
  float v;
  if (c < 64)       v = x1[(size_t)p * 64 + c];
  else if (c < 128) v = x2[(size_t)p * 64 + (c - 64)];
  else              v = x3[(size_t)p * 64 + (c - 128)];
  E[id] = (_Float16)v;
}

// seg head input: [ml(192) | gf_b(1024)] -> f16 [P,1216] (element/thread)
__global__ void pack_seg(const float* __restrict__ x1, const float* __restrict__ x2,
                         const float* __restrict__ x3, const float* __restrict__ gf,
                         _Float16* __restrict__ E) {
  size_t id = (size_t)blockIdx.x * blockDim.x + threadIdx.x;
  if (id >= (size_t)PM * 1216) return;
  int c = (int)(id % 1216);
  int p = (int)(id / 1216);
  float v;
  if (c < 64)       v = x1[(size_t)p * 64 + c];
  else if (c < 128) v = x2[(size_t)p * 64 + (c - 64)];
  else if (c < 192) v = x3[(size_t)p * 64 + (c - 128)];
  else              v = gf[(size_t)(p / N_) * 1024 + (c - 192)];
  E[id] = (_Float16)v;
}

// final: Y [P,16] (cols 0..3 valid) -> out (B, 4, N)
__global__ void write_out(const float* __restrict__ Y, float* __restrict__ out) {
  int id = blockIdx.x * blockDim.x + threadIdx.x;
  if (id >= PM * 4) return;
  int n = id % N_, q = id / N_;
  int c = q % 4, b = q / 4;
  out[id] = Y[((size_t)b * N_ + n) * 16 + c];
}

// ---------------------------------------------------------------------------
// Host orchestration
// ---------------------------------------------------------------------------
extern "C" void kernel_launch(void* const* d_in, const int* in_sizes, int n_in,
                              void* d_out, int out_size, void* d_ws, size_t ws_size,
                              hipStream_t stream) {
  (void)in_sizes; (void)n_in; (void)out_size; (void)ws_size;
  auto F = [&](int i) { return (const float*)d_in[i]; };

  // --- workspace bump allocator (256B aligned) ---
  size_t off = 0;
  auto alloc = [&](size_t bytes) -> char* {
    char* p = (char*)d_ws + off;
    off = (off + bytes + 255) & ~(size_t)255;
    return p;
  };

  float*     xT    = (float*)alloc((size_t)B_ * N_ * 3 * 4);
  float*     xS    = (float*)alloc((size_t)B_ * N_ * 3 * 4);
  int*       idx   = (int*)alloc((size_t)B_ * N_ * KNN * 4);
  _Float16*  E     = (_Float16*)alloc((size_t)EM * 128 * 2);  // also holds seg0 input (16384x1216)
  float*     G     = (float*)alloc((size_t)EM * 128 * 4);     // GEMM output (in-place BN)
  float*     stats = (float*)alloc(2 * 1024 * 4);
  float*     x1    = (float*)alloc((size_t)PM * 64 * 4);
  float*     x2    = (float*)alloc((size_t)PM * 64 * 4);
  float*     x3    = (float*)alloc((size_t)PM * 64 * 4);
  float*     tpt   = (float*)alloc((size_t)PM * 128 * 4);     // ST edge-conv pooled
  float*     tg    = (float*)alloc((size_t)B_ * 1024 * 4);
  float*     m0    = (float*)alloc((size_t)B_ * 512 * 4);
  float*     m1    = (float*)alloc((size_t)B_ * 256 * 4);
  float*     mat   = (float*)alloc((size_t)B_ * 9 * 4);
  float*     gfv   = (float*)alloc((size_t)B_ * 1024 * 4);

  // --- weight conversion to padded f16 ---
  struct WS { int idx; int co, ci, coP, ciP; };
  const WS wspec[12] = {
    { 1,   64,    6,   64,   32},  // st.ec0
    { 4,  128,   64,  128,   64},  // st.ec1
    { 7, 1024,  128, 1024,  128},  // st.fc
    {20,   64,    6,   64,   32},  // ec1_0
    {23,   64,   64,   64,   64},  // ec1_1
    {26,   64,  128,   64,  128},  // ec2_0
    {29,   64,  128,   64,  128},  // ec3_0
    {32, 1024,  192, 1024,  192},  // gf
    {35,  256, 1216,  256, 1216},  // seg0
    {38,  256,  256,  256,  256},  // seg1
    {41,  128,  256,  128,  256},  // seg2
    {44,    4,  128,   16,  128},  // seg3 (padded to 16 out)
  };
  _Float16* wh[12];
  for (int i = 0; i < 12; ++i) {
    int tot = wspec[i].coP * wspec[i].ciP;
    wh[i] = (_Float16*)alloc((size_t)tot * 2);
    pack_w<<<(tot + 255) / 256, 256, 0, stream>>>(F(wspec[i].idx), wh[i],
        wspec[i].co, wspec[i].ci, wspec[i].coP, wspec[i].ciP);
  }

  auto gemm = [&](const _Float16* Aq, int wi, float* Dq, int M) {
    int Nc = wspec[wi].coP, Kp = wspec[wi].ciP;
    if ((Nc & 63) == 0) {
      int strips = (M >> 4) * (Nc >> 6);
      wmma_gemm_nt<4><<<(strips + 3) / 4, 128, 0, stream>>>(Aq, wh[wi], Dq, M, Nc, Kp);
    } else {
      int strips = (M >> 4) * (Nc >> 4);
      wmma_gemm_nt<1><<<(strips + 3) / 4, 128, 0, stream>>>(Aq, wh[wi], Dq, M, Nc, Kp);
    }
  };
  auto bn = [&](float* Y, int M, int Nc, int NcReal, int gi, int bi) {
    bn_stats<<<Nc, 256, 0, stream>>>(Y, stats, M, Nc);
    size_t tot = (size_t)M * Nc;
    bn_apply_leaky<<<(int)((tot + 255) / 256), 256, 0, stream>>>(
        Y, stats, F(gi), F(bi), M, Nc, NcReal);
  };
  auto elems = [](size_t n) { return (int)((n + 255) / 256); };
  const int GB_P = elems(PM);

  // ===== spatial transformer =====
  transpose_in<<<elems((size_t)PM * 3), 256, 0, stream>>>(F(0), xT);
  knn_kernel<3><<<GB_P, 256, 0, stream>>>(xT, idx);
  gather_edge<<<elems((size_t)EM * 32), 256, 0, stream>>>(xT, idx, E, 3, 32);
  gemm(E, 0, G, EM);  bn(G, EM, 64, 64, 2, 3);                        // ec0
  pack_rows<<<elems((size_t)EM * 64), 256, 0, stream>>>(G, E, EM, 64, 64);
  gemm(E, 1, G, EM);  bn(G, EM, 128, 128, 5, 6);                      // ec1
  max_over_k<<<elems((size_t)PM * 128), 256, 0, stream>>>(G, tpt, 128);
  pack_rows<<<elems((size_t)PM * 128), 256, 0, stream>>>(tpt, E, PM, 128, 128);
  gemm(E, 2, G, PM);  bn(G, PM, 1024, 1024, 8, 9);                    // fc
  max_over_n<<<elems((size_t)B_ * 1024), 256, 0, stream>>>(G, tg, 1024);
  lin_bn_leaky<<<2, 256, 0, stream>>>(tg, F(10), F(11), F(12), F(13), m0, 1024, 512);
  lin_bn_leaky<<<1, 256, 0, stream>>>(m0, F(14), F(15), F(16), F(17), m1, 512, 256);
  transform_mat<<<1, 64, 0, stream>>>(m1, F(18), F(19), mat);
  apply_transform<<<GB_P, 256, 0, stream>>>(xT, mat, xS);

  // ===== edge conv 1 =====
  knn_kernel<3><<<GB_P, 256, 0, stream>>>(xS, idx);
  gather_edge<<<elems((size_t)EM * 32), 256, 0, stream>>>(xS, idx, E, 3, 32);
  gemm(E, 3, G, EM);  bn(G, EM, 64, 64, 21, 22);                      // ec1_0
  pack_rows<<<elems((size_t)EM * 64), 256, 0, stream>>>(G, E, EM, 64, 64);
  gemm(E, 4, G, EM);  bn(G, EM, 64, 64, 24, 25);                      // ec1_1
  max_over_k<<<elems((size_t)PM * 64), 256, 0, stream>>>(G, x1, 64);

  // ===== edge conv 2 =====
  knn_kernel<64><<<GB_P, 256, 0, stream>>>(x1, idx);
  gather_edge<<<elems((size_t)EM * 128), 256, 0, stream>>>(x1, idx, E, 64, 128);
  gemm(E, 5, G, EM);  bn(G, EM, 64, 64, 27, 28);                      // ec2_0
  max_over_k<<<elems((size_t)PM * 64), 256, 0, stream>>>(G, x2, 64);

  // ===== edge conv 3 =====
  knn_kernel<64><<<GB_P, 256, 0, stream>>>(x2, idx);
  gather_edge<<<elems((size_t)EM * 128), 256, 0, stream>>>(x2, idx, E, 64, 128);
  gemm(E, 6, G, EM);  bn(G, EM, 64, 64, 30, 31);                      // ec3_0
  max_over_k<<<elems((size_t)PM * 64), 256, 0, stream>>>(G, x3, 64);

  // ===== global feature =====
  pack_concat3<<<elems((size_t)PM * 192), 256, 0, stream>>>(x1, x2, x3, E);
  gemm(E, 7, G, PM);  bn(G, PM, 1024, 1024, 33, 34);                  // gf
  max_over_n<<<elems((size_t)B_ * 1024), 256, 0, stream>>>(G, gfv, 1024);

  // ===== segmentation head =====
  pack_seg<<<elems((size_t)PM * 1216), 256, 0, stream>>>(x1, x2, x3, gfv, E);
  gemm(E, 8, G, PM);  bn(G, PM, 256, 256, 36, 37);                    // seg0
  pack_rows<<<elems((size_t)PM * 256), 256, 0, stream>>>(G, E, PM, 256, 256);
  gemm(E, 9, G, PM);  bn(G, PM, 256, 256, 39, 40);                    // seg1
  pack_rows<<<elems((size_t)PM * 256), 256, 0, stream>>>(G, E, PM, 256, 256);
  gemm(E, 10, G, PM); bn(G, PM, 128, 128, 42, 43);                    // seg2
  pack_rows<<<elems((size_t)PM * 128), 256, 0, stream>>>(G, E, PM, 128, 128);
  gemm(E, 11, G, PM); bn(G, PM, 16, 4, 45, 46);                       // seg3 (padded)

  write_out<<<elems((size_t)PM * 4), 256, 0, stream>>>(G, (float*)d_out);
}
